// NSA_91173565759558
// MI455X (gfx1250) — matrix operand order repaired and checked by reading, not compiled, not run on hardware
//
#include <hip/hip_runtime.h>

typedef __attribute__((ext_vector_type(16))) __bf16 v16bf_t;
typedef __attribute__((ext_vector_type(8)))  float  v8f_t;
typedef __attribute__((ext_vector_type(4)))  unsigned int u32x4;
typedef __attribute__((ext_vector_type(8)))  unsigned int u32x8;

static constexpr int kT   = 2048;   // sequence length (== S)
static constexpr int kDM  = 2048;   // d_model
static constexpr int kH   = 32;     // query heads
static constexpr int kKV  = 8;      // kv heads
static constexpr int kD   = 64;     // head dim
static constexpr int kNB  = 32;     // compressed blocks
static constexpr int kBS  = 64;     // block size
static constexpr int kWIN = 512;    // sliding window
#define ATT_SCALE 0.125f            // 1/sqrt(64), folded into Q at pack time

union AFrag { v16bf_t v; unsigned int u[8]; };

// 16-bit A-matrix (16x32) K index for VGPR j, lane group g (ISA 7.12.2)
__device__ __forceinline__ int amapK(int j, int g) {
    return (j < 4) ? (8 * g + 2 * j) : (16 + 8 * g + 2 * (j - 4));
}

// ---- CDNA5 async global->LDS copy (per-lane, ASYNCcnt) ----------------------
__device__ __forceinline__ void async_copy_b128(void* lds_dst, const void* gsrc) {
    unsigned lds_off = (unsigned)(size_t)lds_dst;            // low 32b of generic = LDS offset
    unsigned long long ga = (unsigned long long)(size_t)gsrc;
    asm volatile("global_load_async_to_lds_b128 %0, %1, off"
                 :: "v"(lds_off), "v"(ga) : "memory");
}
__device__ __forceinline__ void wait_async0() {
    asm volatile("s_wait_asynccnt 0x0" ::: "memory");
}

// ---- CDNA5 Tensor Data Mover: 2D tile (bf16) global -> LDS, TENSORcnt ------
// D# group0/group1 per ISA 8.3/8.4. tensor dims == tile dims (no OOB path),
// dim0 contiguous elements, dim1 rows with stride_d0 (elements).
__device__ __forceinline__ void tdm_load_2d_bf16(void* lds_dst, const void* gsrc,
                                                 unsigned tile_d0, unsigned tile_d1,
                                                 unsigned long long stride_d0) {
    unsigned long long ga = (unsigned long long)(size_t)gsrc;
    u32x4 g0;
    g0.x = 0x1u;                                             // count=1, user mode
    g0.y = (unsigned)(size_t)lds_dst;                        // lds_addr
    g0.z = (unsigned)ga;                                     // global_addr[31:0]
    g0.w = (unsigned)((ga >> 32) & 0x01FFFFFFu) | 0x80000000u; // addr[56:32] | type=2
    u32x8 g1;
    g1.s0 = 0x00010000u;                                     // data_size=1 (2 bytes)
    g1.s1 = (tile_d0 & 0xFFFFu) << 16;                       // tensor_dim0[15:0]
    g1.s2 = ((tile_d0 >> 16) & 0xFFFFu) | ((tile_d1 & 0xFFFFu) << 16); // dim0 hi | dim1 lo
    g1.s3 = ((tile_d1 >> 16) & 0xFFFFu) | ((tile_d0 & 0xFFFFu) << 16); // dim1 hi | tile_dim0
    g1.s4 = tile_d1 & 0xFFFFu;                               // tile_dim1 (tile_dim2 = 0)
    g1.s5 = (unsigned)(stride_d0 & 0xFFFFFFFFull);           // dim0_stride[31:0]
    g1.s6 = (unsigned)((stride_d0 >> 32) & 0xFFFFull);       // dim0_stride[47:32]
    g1.s7 = 0u;                                              // dim1_stride unused (2D tile)
    asm volatile("tensor_load_to_lds %0, %1" :: "s"(g0), "s"(g1) : "memory");
}
__device__ __forceinline__ void wait_tensor0() {
    __builtin_amdgcn_s_wait_tensorcnt(0);
}

// ---------------------------------------------------------------- conversions
__global__ void k_cvt_bf16(const float* __restrict__ in, __bf16* __restrict__ out, int n) {
    int i = blockIdx.x * 256 + threadIdx.x;
    if (i < n) out[i] = (__bf16)in[i];
}

// ------------------------------------------------- tiled bf16 WMMA GEMM (f32 C)
// C[M,N] = A[M,K] * B[K,N], A/B row-major bf16. 64x64 macro tile, 8 waves,
// A tile staged by the Tensor Data Mover; B tile staged manually (transpose).
__global__ void __launch_bounds__(256)
k_gemm_bf16(const __bf16* __restrict__ A, const __bf16* __restrict__ B,
            float* __restrict__ C, int M, int N, int K) {
    __shared__ __align__(16) __bf16 As[64][32];
    __shared__ __align__(16) __bf16 Bt[64][32];   // transposed: [n][k]
    const int tid  = threadIdx.x;
    const int lane = tid & 31, wave = tid >> 5;
    const int g = lane >> 4, ln = lane & 15;
    const int mBase = blockIdx.y * 64, nBase = blockIdx.x * 64;
    const int rT = wave & 3;            // A row tile
    const int cT = (wave >> 2) * 2;     // first of two N tiles
    v8f_t acc0 = {0.f,0.f,0.f,0.f,0.f,0.f,0.f,0.f};
    v8f_t acc1 = acc0;
    const int la = tid * 8;
    const int bk = la >> 6, bn = la & 63;   // B tile coords (32x64)
    for (int k0 = 0; k0 < K; k0 += 32) {
        if (wave == 0) {   // one TDM descriptor per k-step: 64x32 bf16 tile, row stride K
            tdm_load_2d_bf16(&As[0][0], &A[(size_t)mBase * K + k0], 32u, 64u,
                             (unsigned long long)K);
        }
        uint4 bd = *(const uint4*)&B[(size_t)(k0 + bk) * N + nBase + bn];
        __bf16 tmp[8]; *(uint4*)tmp = bd;
        #pragma unroll
        for (int e = 0; e < 8; ++e) Bt[bn + e][bk] = tmp[e];
        if (k0 + 32 < K) {  // prefetch next B slab (speculative, gfx1250 global_prefetch)
            __builtin_prefetch(&B[(size_t)(k0 + 32 + bk) * N + nBase + bn], 0, 1);
        }
        if (wave == 0) wait_tensor0();
        __syncthreads();
        AFrag a, b0, b1;
        #pragma unroll
        for (int j = 0; j < 8; ++j) {
            a.u[j]  = *(const unsigned int*)&As[rT * 16 + ln][amapK(j, g)];
            int kb = amapK(j, g);
            b0.u[j] = *(const unsigned int*)&Bt[cT * 16 + ln][kb];
            b1.u[j] = *(const unsigned int*)&Bt[(cT + 1) * 16 + ln][kb];
        }
        acc0 = __builtin_amdgcn_wmma_f32_16x16x32_bf16(false, a.v, false, b0.v, (short)0, acc0, false, false);
        acc1 = __builtin_amdgcn_wmma_f32_16x16x32_bf16(false, a.v, false, b1.v, (short)0, acc1, false, false);
        __syncthreads();
    }
    #pragma unroll
    for (int i = 0; i < 8; ++i) {
        int row = mBase + rT * 16 + i + 8 * g;
        C[(size_t)row * N + nBase + cT * 16 + ln]       = acc0[i];
        C[(size_t)row * N + nBase + (cT + 1) * 16 + ln] = acc1[i];
    }
}

// ------------------------------------------------------------ RoPE + repack
__global__ void k_rope_pack_q(const float* __restrict__ Qf, __bf16* __restrict__ Qb) {
    int idx = blockIdx.x * 256 + threadIdx.x;          // T*H*32 pairs
    if (idx >= kT * kH * 32) return;
    int p = idx & 31, h = (idx >> 5) & 31, t = idx >> 10;
    float inv = __expf(-(float)p * 0.28782313662425572f);  // ln(1e4)/32
    float ang = (float)t * inv;
    float c = cosf(ang), s = sinf(ang);
    const float* src = &Qf[(size_t)t * kDM + h * kD + 2 * p];
    float x0 = src[0], x1 = src[1];
    __bf16* dst = &Qb[((size_t)h * kT + t) * kD + 2 * p];
    dst[0] = (__bf16)((x0 * c - x1 * s) * ATT_SCALE);
    dst[1] = (__bf16)((x0 * s + x1 * c) * ATT_SCALE);
}

__global__ void k_rope_pack_k(const float* __restrict__ Kf, __bf16* __restrict__ Kb) {
    int idx = blockIdx.x * 256 + threadIdx.x;          // T*KV*32 pairs
    if (idx >= kT * kKV * 32) return;
    int p = idx & 31, kv = (idx >> 5) & 7, t = idx >> 8;
    float inv = __expf(-(float)p * 0.28782313662425572f);
    float ang = (float)t * inv;
    float c = cosf(ang), s = sinf(ang);
    const float* src = &Kf[(size_t)t * (kKV * kD) + kv * kD + 2 * p];
    float x0 = src[0], x1 = src[1];
    __bf16* dst = &Kb[((size_t)kv * kT + t) * kD + 2 * p];
    dst[0] = (__bf16)(x0 * c - x1 * s);
    dst[1] = (__bf16)(x0 * s + x1 * c);
}

__global__ void k_pack_v(const float* __restrict__ Vf, __bf16* __restrict__ Vb) {
    int idx = blockIdx.x * 256 + threadIdx.x;          // T*KV*64
    if (idx >= kT * kKV * kD) return;
    int d = idx & 63, kv = (idx >> 6) & 7, t = idx >> 9;
    Vb[((size_t)kv * kT + t) * kD + d] = (__bf16)Vf[(size_t)t * (kKV * kD) + kv * kD + d];
}

// ------------------------------------------------------------------- gates
__global__ void k_gates(const float* __restrict__ x, const float* __restrict__ Wg,
                        float* __restrict__ gates) {
    int t = blockIdx.x * 64 + threadIdx.x;
    if (t >= kT) return;
    float a0 = 0.f, a1 = 0.f, a2 = 0.f;
    const float* xr = &x[(size_t)t * kDM];
    for (int i = 0; i < kDM; ++i) {
        float xv = xr[i];
        a0 += xv * Wg[i * 3 + 0];
        a1 += xv * Wg[i * 3 + 1];
        a2 += xv * Wg[i * 3 + 2];
    }
    float m = fmaxf(a0, fmaxf(a1, a2));
    float e0 = __expf(a0 - m), e1 = __expf(a1 - m), e2 = __expf(a2 - m);
    float inv = 1.f / (e0 + e1 + e2);
    gates[t * 3 + 0] = e0 * inv;
    gates[t * 3 + 1] = e1 * inv;
    gates[t * 3 + 2] = e2 * inv;
}

// --------------------------------------------------- block compression GEMMs
__global__ void k_compress(const __bf16* __restrict__ KV, const float* __restrict__ W,
                           float* __restrict__ out) {
    int n = blockIdx.x, kv = blockIdx.y, d = threadIdx.x;
    const __bf16* base = &KV[((size_t)kv * kT + n * kBS) * kD];  // contiguous 4096
    float acc = 0.f;
    for (int i = 0; i < kBS * kD; ++i)
        acc += (float)base[i] * W[(size_t)i * kD + d];
    out[((size_t)kv * kNB + n) * kD + d] = acc;
}

// ------------------------------------------- compressed-branch attention (nb=32)
__global__ void k_attn_compressed(const __bf16* __restrict__ Qb, const float* __restrict__ kc,
                                  const float* __restrict__ vc, const float* __restrict__ gates,
                                  float* __restrict__ ctx) {
    int idx = blockIdx.x * 256 + threadIdx.x;   // H*T threads
    int t = idx & (kT - 1);
    int h = idx >> 11;
    int kv = h >> 2;
    float q[kD];
    const __bf16* qp = &Qb[((size_t)h * kT + t) * kD];
    #pragma unroll
    for (int d = 0; d < kD; ++d) q[d] = (float)qp[d];   // pre-scaled
    float lg[kNB], m = -1e30f;
    const float* kcb = &kc[(size_t)kv * kNB * kD];
    for (int n = 0; n < kNB; ++n) {
        float acc = 0.f;
        const float* kr = &kcb[n * kD];
        #pragma unroll 8
        for (int d = 0; d < kD; ++d) acc += q[d] * kr[d];
        lg[n] = acc; m = fmaxf(m, acc);
    }
    float ssum = 0.f;
    for (int n = 0; n < kNB; ++n) { lg[n] = __expf(lg[n] - m); ssum += lg[n]; }
    float w = gates[t * 3 + 0] / ssum;
    const float* vcb = &vc[(size_t)kv * kNB * kD];
    float* op = &ctx[(size_t)t * kDM + h * kD];
    for (int d = 0; d < kD; ++d) {
        float o = 0.f;
        for (int n = 0; n < kNB; ++n) o += lg[n] * vcb[n * kD + d];
        op[d] = w * o;                      // initializes ctx
    }
}

// ---------------------------- flash attention (dense or windowed) with WMMA
// grid (T/128, H); 8 waves, one 16-row query strip per wave; S tiles of 32.
// K tile staged via async global->LDS (ASYNCcnt); V tile staged transposed.
__global__ void __launch_bounds__(256)
k_attn(const __bf16* __restrict__ Qb, const __bf16* __restrict__ Kb,
       const __bf16* __restrict__ Vb, const float* __restrict__ gates,
       float* __restrict__ ctx, int gate_idx, int windowed) {
    __shared__ __align__(16) __bf16 Kt[32][64];       // [s_local][d]
    __shared__ __align__(16) __bf16 Vt[64][32];       // [d][s_local] transposed
    __shared__ __align__(16) __bf16 Pb[8][16][32];    // per-wave P tile
    const int tid  = threadIdx.x;
    const int lane = tid & 31, wave = tid >> 5;
    const int g = lane >> 4, ln = lane & 15;
    const int h = blockIdx.y, kv = h >> 2;
    const int qbase = blockIdx.x * 128;
    const int t0 = qbase + wave * 16;
    AFrag a0, a1;
    const __bf16* qrow = &Qb[((size_t)h * kT + t0 + ln) * kD];
    #pragma unroll
    for (int j = 0; j < 8; ++j) {
        int K0 = 16 * g + 2 * j;
        a0.u[j] = *(const unsigned int*)&qrow[K0];
        a1.u[j] = *(const unsigned int*)&qrow[K0 + 32];
    }
    float M[8], L[8];
    v8f_t o[4];
    #pragma unroll
    for (int i = 0; i < 8; ++i) { M[i] = -1e30f; L[i] = 0.f; }
    #pragma unroll
    for (int dt = 0; dt < 4; ++dt) o[dt] = (v8f_t){0.f,0.f,0.f,0.f,0.f,0.f,0.f,0.f};

    int s_lo = 0, s_hi = kT;
    if (windowed) {
        int lo = qbase - (kWIN - 1);
        s_lo = lo > 0 ? (lo & ~31) : 0;
        s_hi = qbase + 128;
    }
    const int la = tid * 8;
    const int kr = la >> 6, lc = la & 63;   // 32x64 staging coords
    for (int sbase = s_lo; sbase < s_hi; sbase += 32) {
        // K tile: per-lane async DMA to LDS (no VGPR round trip)
        async_copy_b128(&Kt[kr][lc], &Kb[((size_t)kv * kT + sbase + kr) * kD + lc]);
        // V tile: load + transpose into LDS
        uint4 vd = *(const uint4*)&Vb[((size_t)kv * kT + sbase + kr) * kD + lc];
        __bf16 vt[8]; *(uint4*)vt = vd;
        #pragma unroll
        for (int e = 0; e < 8; ++e) Vt[lc + e][kr] = vt[e];
        if (sbase + 32 < s_hi) {   // prefetch next S tile
            __builtin_prefetch(&Kb[((size_t)kv * kT + sbase + 32 + kr) * kD + lc], 0, 1);
            __builtin_prefetch(&Vb[((size_t)kv * kT + sbase + 32 + kr) * kD + lc], 0, 1);
        }
        wait_async0();
        __syncthreads();
        bool active = !windowed || ((sbase <= t0 + 15) && (sbase + 31 >= t0 - (kWIN - 1)));
        if (active) {  // uniform per wave -> EXEC all-ones for WMMA
            v8f_t c0 = (v8f_t){0.f,0.f,0.f,0.f,0.f,0.f,0.f,0.f}, c1 = c0;
            AFrag b;
            #pragma unroll
            for (int ks = 0; ks < 2; ++ks) {
                #pragma unroll
                for (int j = 0; j < 8; ++j)
                    b.u[j] = *(const unsigned int*)&Kt[ln][ks * 32 + 16 * g + 2 * j];
                c0 = __builtin_amdgcn_wmma_f32_16x16x32_bf16(false, ks ? a1.v : a0.v, false, b.v, (short)0, c0, false, false);
                #pragma unroll
                for (int j = 0; j < 8; ++j)
                    b.u[j] = *(const unsigned int*)&Kt[16 + ln][ks * 32 + 16 * g + 2 * j];
                c1 = __builtin_amdgcn_wmma_f32_16x16x32_bf16(false, ks ? a1.v : a0.v, false, b.v, (short)0, c1, false, false);
            }
            #pragma unroll
            for (int i = 0; i < 8; ++i) {
                int trow = t0 + i + 8 * g;
                float v0 = c0[i], v1 = c1[i];
                if (windowed) {
                    int s0 = sbase + ln, s1 = sbase + 16 + ln;
                    if (s0 > trow || s0 < trow - (kWIN - 1)) v0 = -1e30f;
                    if (s1 > trow || s1 < trow - (kWIN - 1)) v1 = -1e30f;
                }
                float mx = fmaxf(v0, v1);
                #pragma unroll
                for (int dlt = 1; dlt < 16; dlt <<= 1)
                    mx = fmaxf(mx, __shfl_xor(mx, dlt, 32));
                float nm = fmaxf(M[i], mx);
                float p0 = (v0 <= -1e29f) ? 0.f : __expf(v0 - nm);
                float p1 = (v1 <= -1e29f) ? 0.f : __expf(v1 - nm);
                float rs = p0 + p1;
                #pragma unroll
                for (int dlt = 1; dlt < 16; dlt <<= 1)
                    rs += __shfl_xor(rs, dlt, 32);
                float al = __expf(M[i] - nm);
                L[i] = L[i] * al + rs;
                M[i] = nm;
                #pragma unroll
                for (int dt = 0; dt < 4; ++dt) o[dt][i] *= al;
                Pb[wave][i + 8 * g][ln]      = (__bf16)p0;
                Pb[wave][i + 8 * g][16 + ln] = (__bf16)p1;
            }
            AFrag pa, vb;
            #pragma unroll
            for (int j = 0; j < 8; ++j)
                pa.u[j] = *(const unsigned int*)&Pb[wave][ln][16 * g + 2 * j];
            #pragma unroll
            for (int dt = 0; dt < 4; ++dt) {
                #pragma unroll
                for (int j = 0; j < 8; ++j)
                    vb.u[j] = *(const unsigned int*)&Vt[dt * 16 + ln][16 * g + 2 * j];
                o[dt] = __builtin_amdgcn_wmma_f32_16x16x32_bf16(false, pa.v, false, vb.v, (short)0, o[dt], false, false);
            }
        }
        __syncthreads();
    }
    #pragma unroll
    for (int i = 0; i < 8; ++i) {
        int trow = t0 + i + 8 * g;
        float inv = (L[i] > 0.f) ? 1.f / L[i] : 0.f;   // nan_to_num semantics
        float w = gates[trow * 3 + gate_idx] * inv;
        float* dst = &ctx[(size_t)trow * kDM + h * kD];
        #pragma unroll
        for (int dt = 0; dt < 4; ++dt)
            dst[dt * 16 + ln] += w * o[dt][i];
    }
}

// =============================================================== launcher
extern "C" void kernel_launch(void* const* d_in, const int* in_sizes, int n_in,
                              void* d_out, int out_size, void* d_ws, size_t ws_size,
                              hipStream_t stream) {
    (void)in_sizes; (void)n_in; (void)out_size; (void)ws_size;
    const float* x   = (const float*)d_in[0];
    const float* Wq  = (const float*)d_in[1];
    const float* Wk  = (const float*)d_in[2];
    const float* Wv  = (const float*)d_in[3];
    const float* Wo  = (const float*)d_in[4];
    const float* Wck = (const float*)d_in[5];
    const float* Wcv = (const float*)d_in[6];
    /* d_in[7] = W_imp -> dead code in reference (top_k discarded) */
    const float* Wg  = (const float*)d_in[8];

    char* ws = (char*)d_ws;
    size_t off = 0;
    auto alloc = [&](size_t bytes) -> void* {
        void* p = ws + off;
        off = (off + bytes + 255) & ~(size_t)255;
        return p;
    };
    __bf16* Xb   = (__bf16*)alloc((size_t)kT * kDM * 2);
    __bf16* Wqb  = (__bf16*)alloc((size_t)kDM * (kH * kD) * 2);
    __bf16* Wkb  = (__bf16*)alloc((size_t)kDM * (kKV * kD) * 2);
    __bf16* Wvb  = (__bf16*)alloc((size_t)kDM * (kKV * kD) * 2);
    __bf16* Wob  = (__bf16*)alloc((size_t)(kH * kD) * kDM * 2);
    float*  Qf   = (float*) alloc((size_t)kT * (kH * kD) * 4);
    float*  Kf   = (float*) alloc((size_t)kT * (kKV * kD) * 4);
    float*  Vf   = (float*) alloc((size_t)kT * (kKV * kD) * 4);
    __bf16* Qb   = (__bf16*)alloc((size_t)kH * kT * kD * 2);
    __bf16* Kb   = (__bf16*)alloc((size_t)kKV * kT * kD * 2);
    __bf16* Vb   = (__bf16*)alloc((size_t)kKV * kT * kD * 2);
    float*  kcf  = (float*) alloc((size_t)kKV * kNB * kD * 4);
    float*  vcf  = (float*) alloc((size_t)kKV * kNB * kD * 4);
    float*  gts  = (float*) alloc((size_t)kT * 3 * 4);
    float*  ctx  = (float*) alloc((size_t)kT * kDM * 4);
    __bf16* comb = (__bf16*)alloc((size_t)kT * kDM * 2);

    int n;
    n = kT * kDM;             k_cvt_bf16<<<(n + 255) / 256, 256, 0, stream>>>(x,  Xb,  n);
    n = kDM * kH * kD;        k_cvt_bf16<<<(n + 255) / 256, 256, 0, stream>>>(Wq, Wqb, n);
    n = kDM * kKV * kD;       k_cvt_bf16<<<(n + 255) / 256, 256, 0, stream>>>(Wk, Wkb, n);
    n = kDM * kKV * kD;       k_cvt_bf16<<<(n + 255) / 256, 256, 0, stream>>>(Wv, Wvb, n);
    n = kH * kD * kDM;        k_cvt_bf16<<<(n + 255) / 256, 256, 0, stream>>>(Wo, Wob, n);

    // projections
    k_gemm_bf16<<<dim3((kH * kD) / 64, kT / 64),  256, 0, stream>>>(Xb, Wqb, Qf, kT, kH * kD,  kDM);
    k_gemm_bf16<<<dim3((kKV * kD) / 64, kT / 64), 256, 0, stream>>>(Xb, Wkb, Kf, kT, kKV * kD, kDM);
    k_gemm_bf16<<<dim3((kKV * kD) / 64, kT / 64), 256, 0, stream>>>(Xb, Wvb, Vf, kT, kKV * kD, kDM);

    // RoPE + head-major bf16 repack
    n = kT * kH * 32;   k_rope_pack_q<<<(n + 255) / 256, 256, 0, stream>>>(Qf, Qb);
    n = kT * kKV * 32;  k_rope_pack_k<<<(n + 255) / 256, 256, 0, stream>>>(Kf, Kb);
    n = kT * kKV * kD;  k_pack_v<<<(n + 255) / 256, 256, 0, stream>>>(Vf, Vb);

    k_gates<<<kT / 64, 64, 0, stream>>>(x, Wg, gts);

    // compressed K/V blocks
    k_compress<<<dim3(kNB, kKV), 64, 0, stream>>>(Kb, Wck, kcf);
    k_compress<<<dim3(kNB, kKV), 64, 0, stream>>>(Vb, Wcv, vcf);

    // branch 1: compressed attention (initializes ctx with gate 0)
    k_attn_compressed<<<(kH * kT) / 256, 256, 0, stream>>>(Qb, kcf, vcf, gts, ctx);
    // branch 2: dense "selected" attention (gate 1)
    k_attn<<<dim3(kT / 128, kH), 256, 0, stream>>>(Qb, Kb, Vb, gts, ctx, 1, 0);
    // branch 3: sliding-window attention (gate 2)
    k_attn<<<dim3(kT / 128, kH), 256, 0, stream>>>(Qb, Kb, Vb, gts, ctx, 2, 1);

    // combine -> bf16 -> output projection
    n = kT * kDM;  k_cvt_bf16<<<(n + 255) / 256, 256, 0, stream>>>(ctx, comb, n);
    k_gemm_bf16<<<dim3(kDM / 64, kT / 64), 256, 0, stream>>>(comb, Wob, (float*)d_out, kT, kDM, kH * kD);
}
